// GNNStack_33827162423505
// MI455X (gfx1250) — matrix operand order repaired and checked by reading, not compiled, not run on hardware
//
#include <hip/hip_runtime.h>

typedef __attribute__((ext_vector_type(16))) __bf16 v16bf;
typedef __attribute__((ext_vector_type(8)))  __bf16 v8bf;
typedef __attribute__((ext_vector_type(8)))  float  v8f;

#define NN 100000
#define NE 600000

// ---------------- WMMA fragment helpers (CDNA5 wave32 layouts) ----------------
// A (bf16 16x32): lane L: row = L&15; lanes 0-15 hold K {kk+0..7, kk+16..23},
// lanes 16-31 hold K {kk+8..15, kk+24..31}. Two contiguous 16B LDS chunks.
__device__ __forceinline__ v16bf load_a_frag(const __bf16* rowp, int kk, int khalf) {
  v8bf lo = *(const v8bf*)(rowp + kk + khalf);
  v8bf hi = *(const v8bf*)(rowp + kk + 16 + khalf);
  return __builtin_shufflevector(lo, hi, 0,1,2,3,4,5,6,7,8,9,10,11,12,13,14,15);
}

__device__ __forceinline__ v8f wmma_bf16(v16bf a, v16bf b, v8f c) {
  return __builtin_amdgcn_wmma_f32_16x16x32_bf16(false, a, false, b, (short)0, c, false, false);
}

__device__ __forceinline__ void atomic_add_f32(float* p, float v) {
  unsafeAtomicAdd(p, v);  // -> global_atomic_add_f32
}

// ---------------- weight repack: f32 [rows,64] -> per-lane B-fragment order ---
// out[((kb*4 + nb)*32 + lane)*16 + i] = W[(kb*32 + (lane<16?0:8) + (i&7) + (i>>3)*16)*64 + nb*16 + (lane&15)]
__global__ void pack_w_kernel(const float* __restrict__ W, __bf16* __restrict__ out,
                              int rows, int KP) {
  int t = blockIdx.x * 256 + threadIdx.x;
  if (t >= KP * 64) return;
  int i    = t & 15;
  int lane = (t >> 4) & 31;
  int nb   = (t >> 9) & 3;
  int kb   = t >> 11;
  int r = kb * 32 + ((lane < 16) ? 0 : 8) + (i & 7) + ((i >> 3) << 4);
  int c = nb * 16 + (lane & 15);
  out[t] = (r < rows) ? (__bf16)W[r * 64 + c] : (__bf16)0.0f;
}

__global__ void zero_kernel(float* p, int n) {
  int t = blockIdx.x * 256 + threadIdx.x;
  if (t < n) p[t] = 0.0f;
}

__global__ void deg_kernel(const int* __restrict__ ei, float* __restrict__ deg, int n_edges) {
  int t = blockIdx.x * 256 + threadIdx.x;
  if (t < n_edges) atomic_add_f32(&deg[ei[n_edges + t]], 1.0f);
}

// ---------------- message + scatter-mean numerator --------------------------
// m = relu([x[src] | edge_attr] @ W + b);  agg[dst] += m   (16 edges per wave)
template<int KP>
__global__ __launch_bounds__(256)
void msg_agg_kernel(const float* __restrict__ x,
                    const float* __restrict__ ea32, const __bf16* __restrict__ ea16,
                    int ea_dim, const int* __restrict__ ei,
                    const v16bf* __restrict__ wp, const float* __restrict__ bias,
                    float* __restrict__ agg, int n_edges) {
  __shared__ __align__(16) __bf16 smem[8][16][KP];
  int lane = threadIdx.x & 31, wave = threadIdx.x >> 5;
  int tile = blockIdx.x * 8 + wave;
  bool active = tile < (n_edges >> 4);
  __bf16 (*A)[KP] = smem[wave];
  if (active) {
    int e0 = tile * 16;
    for (int t = lane; t < 16 * KP; t += 32) {
      int r = t / KP, c = t % KP, e = e0 + r;
      float v = 0.0f;
      if (c < 64)                 v = x[ei[e] * 64 + c];
      else if (c < 64 + ea_dim) { int cc = c - 64;
                                  v = ea32 ? ea32[e * ea_dim + cc] : (float)ea16[e * ea_dim + cc]; }
      A[r][c] = (__bf16)v;
    }
  }
  __syncthreads();
  if (!active) return;
  v8f z = {0.f,0.f,0.f,0.f,0.f,0.f,0.f,0.f};
  v8f acc[4] = {z, z, z, z};
  int row = lane & 15, khalf = (lane < 16) ? 0 : 8;
  #pragma unroll
  for (int kb = 0; kb < KP / 32; ++kb) {
    v16bf a = load_a_frag(&A[row][0], kb * 32, khalf);
    #pragma unroll
    for (int nb = 0; nb < 4; ++nb)
      acc[nb] = wmma_bf16(a, wp[(kb * 4 + nb) * 32 + lane], acc[nb]);
  }
  int e0 = tile * 16, rowoff = (lane >> 4) << 3, col0 = lane & 15;
  int dsts[8];
  #pragma unroll
  for (int r = 0; r < 8; ++r) dsts[r] = ei[n_edges + e0 + rowoff + r];
  #pragma unroll
  for (int nb = 0; nb < 4; ++nb) {
    int col = nb * 16 + col0;
    float bv = bias[col];
    #pragma unroll
    for (int r = 0; r < 8; ++r) {
      float v = acc[nb][r] + bv;
      v = v > 0.f ? v : 0.f;
      atomic_add_f32(&agg[dsts[r] * 64 + col], v);
    }
  }
}

// ---------------- node update: x = l2norm(relu([agg/deg | x] @ W + b)) ------
__global__ __launch_bounds__(256)
void agg_update_kernel(const float* __restrict__ agg, const float* __restrict__ deg,
                       const float* __restrict__ xold, const v16bf* __restrict__ wp,
                       const float* __restrict__ bias, float* __restrict__ xnew,
                       int n_nodes) {
  __shared__ __align__(16) __bf16 smem[8][16][128];   // reused as f32 [16][64]
  __shared__ float Ns[8][16];
  int lane = threadIdx.x & 31, wave = threadIdx.x >> 5;
  int tile = blockIdx.x * 8 + wave;
  bool active = tile < (n_nodes >> 4);
  __bf16 (*A)[128] = smem[wave];
  float* Os = (float*)&smem[wave][0][0];
  if (active) {
    int n0 = tile * 16;
    for (int t = lane; t < 16 * 128; t += 32) {
      int r = t >> 7, c = t & 127, n = n0 + r;
      float v;
      if (c < 64) v = agg[n * 64 + c] * (1.0f / fmaxf(deg[n], 1.0f));
      else        v = xold[n * 64 + (c - 64)];
      A[r][c] = (__bf16)v;
    }
  }
  __syncthreads();
  v8f z = {0.f,0.f,0.f,0.f,0.f,0.f,0.f,0.f};
  v8f acc[4] = {z, z, z, z};
  if (active) {
    int row = lane & 15, khalf = (lane < 16) ? 0 : 8;
    #pragma unroll
    for (int kb = 0; kb < 4; ++kb) {
      v16bf a = load_a_frag(&A[row][0], kb * 32, khalf);
      #pragma unroll
      for (int nb = 0; nb < 4; ++nb)
        acc[nb] = wmma_bf16(a, wp[(kb * 4 + nb) * 32 + lane], acc[nb]);
    }
  }
  __syncthreads();                       // A tile consumed; safe to overwrite
  if (active) {
    int rowoff = (lane >> 4) << 3, col0 = lane & 15;
    #pragma unroll
    for (int nb = 0; nb < 4; ++nb) {
      int col = nb * 16 + col0;
      float bv = bias[col];
      #pragma unroll
      for (int r = 0; r < 8; ++r) {
        float v = acc[nb][r] + bv;
        Os[(rowoff + r) * 64 + col] = v > 0.f ? v : 0.f;
      }
    }
  }
  __syncthreads();
  if (active && lane < 16) {
    float s = 0.f;
    for (int c = 0; c < 64; ++c) { float v = Os[lane * 64 + c]; s += v * v; }
    Ns[wave][lane] = 1.0f / fmaxf(sqrtf(s), 1e-12f);
  }
  __syncthreads();
  if (active) {
    int n0 = tile * 16;
    for (int t = lane; t < 16 * 64; t += 32) {
      int r = t >> 6, c = t & 63;
      xnew[(n0 + r) * 64 + c] = Os[r * 64 + c] * Ns[wave][r];
    }
  }
}

// ---------------- edge update: e' = relu([x[src]|x[dst]|e] @ W + b), bf16 out
template<int KP>
__global__ __launch_bounds__(256)
void edge_mlp_kernel(const float* __restrict__ x,
                     const float* __restrict__ ea32, const __bf16* __restrict__ ea16,
                     int ea_dim, const int* __restrict__ ei,
                     const v16bf* __restrict__ wp, const float* __restrict__ bias,
                     __bf16* __restrict__ eout, int n_edges) {
  __shared__ __align__(16) __bf16 smem[8][16][KP];
  int lane = threadIdx.x & 31, wave = threadIdx.x >> 5;
  int tile = blockIdx.x * 8 + wave;
  bool active = tile < (n_edges >> 4);
  __bf16 (*A)[KP] = smem[wave];
  if (active) {
    int e0 = tile * 16;
    for (int t = lane; t < 16 * KP; t += 32) {
      int r = t / KP, c = t % KP, e = e0 + r;
      float v = 0.0f;
      if (c < 64)                  v = x[ei[e] * 64 + c];
      else if (c < 128)            v = x[ei[n_edges + e] * 64 + (c - 64)];
      else if (c < 128 + ea_dim) { int cc = c - 128;
                                   v = ea32 ? ea32[e * ea_dim + cc] : (float)ea16[e * ea_dim + cc]; }
      A[r][c] = (__bf16)v;
    }
  }
  __syncthreads();
  if (!active) return;
  v8f z = {0.f,0.f,0.f,0.f,0.f,0.f,0.f,0.f};
  v8f acc[4] = {z, z, z, z};
  int row = lane & 15, khalf = (lane < 16) ? 0 : 8;
  #pragma unroll
  for (int kb = 0; kb < KP / 32; ++kb) {
    v16bf a = load_a_frag(&A[row][0], kb * 32, khalf);
    #pragma unroll
    for (int nb = 0; nb < 4; ++nb)
      acc[nb] = wmma_bf16(a, wp[(kb * 4 + nb) * 32 + lane], acc[nb]);
  }
  int e0 = tile * 16, rowoff = (lane >> 4) << 3, col0 = lane & 15;
  #pragma unroll
  for (int nb = 0; nb < 4; ++nb) {
    int col = nb * 16 + col0;
    float bv = bias[col];
    #pragma unroll
    for (int r = 0; r < 8; ++r) {
      float v = acc[nb][r] + bv;
      v = v > 0.f ? v : 0.f;
      eout[(e0 + rowoff + r) * 64 + col] = (__bf16)v;
    }
  }
}

// ---------------- post MLP: out = relu(x@W0+b0)@W1 + b1 ----------------------
__global__ __launch_bounds__(256)
void post_kernel(const float* __restrict__ x,
                 const v16bf* __restrict__ wp0, const float* __restrict__ b0,
                 const v16bf* __restrict__ wp1, const float* __restrict__ b1,
                 float* __restrict__ out, int n_nodes) {
  __shared__ __align__(16) __bf16 As[8][16][64];
  __shared__ __align__(16) __bf16 Hs[8][16][64];
  int lane = threadIdx.x & 31, wave = threadIdx.x >> 5;
  int tile = blockIdx.x * 8 + wave;
  bool active = tile < (n_nodes >> 4);
  int row = lane & 15, khalf = (lane < 16) ? 0 : 8;
  int rowoff = (lane >> 4) << 3, col0 = lane & 15;
  v8f z = {0.f,0.f,0.f,0.f,0.f,0.f,0.f,0.f};
  if (active) {
    int n0 = tile * 16;
    for (int t = lane; t < 16 * 64; t += 32)
      As[wave][t >> 6][t & 63] = (__bf16)x[(n0 + (t >> 6)) * 64 + (t & 63)];
  }
  __syncthreads();
  v8f acc[4] = {z, z, z, z};
  if (active) {
    #pragma unroll
    for (int kb = 0; kb < 2; ++kb) {
      v16bf a = load_a_frag(&As[wave][row][0], kb * 32, khalf);
      #pragma unroll
      for (int nb = 0; nb < 4; ++nb)
        acc[nb] = wmma_bf16(a, wp0[(kb * 4 + nb) * 32 + lane], acc[nb]);
    }
    #pragma unroll
    for (int nb = 0; nb < 4; ++nb) {
      int col = nb * 16 + col0;
      float bv = b0[col];
      #pragma unroll
      for (int r = 0; r < 8; ++r) {
        float v = acc[nb][r] + bv;
        Hs[wave][rowoff + r][col] = (__bf16)(v > 0.f ? v : 0.f);
      }
    }
  }
  __syncthreads();
  if (!active) return;
  v8f acc2[4] = {z, z, z, z};
  #pragma unroll
  for (int kb = 0; kb < 2; ++kb) {
    v16bf a = load_a_frag(&Hs[wave][row][0], kb * 32, khalf);
    #pragma unroll
    for (int nb = 0; nb < 4; ++nb)
      acc2[nb] = wmma_bf16(a, wp1[(kb * 4 + nb) * 32 + lane], acc2[nb]);
  }
  int n0 = tile * 16;
  #pragma unroll
  for (int nb = 0; nb < 4; ++nb) {
    int col = nb * 16 + col0;
    float bv = b1[col];
    #pragma unroll
    for (int r = 0; r < 8; ++r)
      out[(n0 + rowoff + r) * 64 + col] = acc2[nb][r] + bv;
  }
}

// =============================== host side ==================================
extern "C" void kernel_launch(void* const* d_in, const int* in_sizes, int n_in,
                              void* d_out, int out_size, void* d_ws, size_t ws_size,
                              hipStream_t stream) {
  const float* x_in  = (const float*)d_in[0];
  const float* ea_in = (const float*)d_in[1];
  const int*   ei    = (const int*)d_in[2];

  const float *msgW[3], *msgB[3], *aggW[3], *aggB[3], *emlW[3], *emlB[3], *postW[2], *postB[2];
  if (n_in >= 25) {   // each list element is its own input
    for (int l = 0; l < 3; ++l) {
      msgW[l] = (const float*)d_in[3 + l];   msgB[l] = (const float*)d_in[6 + l];
      aggW[l] = (const float*)d_in[9 + l];   aggB[l] = (const float*)d_in[12 + l];
      emlW[l] = (const float*)d_in[15 + l];  emlB[l] = (const float*)d_in[18 + l];
    }
    postW[0] = (const float*)d_in[21]; postW[1] = (const float*)d_in[22];
    postB[0] = (const float*)d_in[23]; postB[1] = (const float*)d_in[24];
  } else {            // lists concatenated flat
    const float* p;
    p = (const float*)d_in[3];  msgW[0] = p; msgW[1] = p + 80*64;  msgW[2] = p + (80+128)*64;
    p = (const float*)d_in[4];  msgB[0] = p; msgB[1] = p + 64;     msgB[2] = p + 128;
    p = (const float*)d_in[5];  aggW[0] = p; aggW[1] = p + 128*64; aggW[2] = p + 256*64;
    p = (const float*)d_in[6];  aggB[0] = p; aggB[1] = p + 64;     aggB[2] = p + 128;
    p = (const float*)d_in[7];  emlW[0] = p; emlW[1] = p + 144*64; emlW[2] = p + (144+192)*64;
    p = (const float*)d_in[8];  emlB[0] = p; emlB[1] = p + 64;     emlB[2] = p + 128;
    p = (const float*)d_in[9];  postW[0] = p; postW[1] = p + 64*64;
    p = (const float*)d_in[10]; postB[0] = p; postB[1] = p + 64;
  }

  char* ws = (char*)d_ws;
  size_t off = 0;
  auto alloc = [&](size_t bytes) -> void* {
    void* r = ws + off;
    off = (off + bytes + 255) & ~(size_t)255;
    return r;
  };
  float*  deg  = (float*)alloc((size_t)NN * 4);
  float*  agg  = (float*)alloc((size_t)NN * 64 * 4);
  float*  x0   = (float*)alloc((size_t)NN * 64 * 4);
  float*  x1   = (float*)alloc((size_t)NN * 64 * 4);
  __bf16* e0b  = (__bf16*)alloc((size_t)NE * 64 * 2);
  __bf16* e1b  = (__bf16*)alloc((size_t)NE * 64 * 2);
  __bf16* msgP[3] = { (__bf16*)alloc(96*64*2),  (__bf16*)alloc(128*64*2), (__bf16*)alloc(128*64*2) };
  __bf16* aggP[3] = { (__bf16*)alloc(128*64*2), (__bf16*)alloc(128*64*2), (__bf16*)alloc(128*64*2) };
  __bf16* emlP[2] = { (__bf16*)alloc(160*64*2), (__bf16*)alloc(192*64*2) };
  __bf16* postP[2]= { (__bf16*)alloc(64*64*2),  (__bf16*)alloc(64*64*2) };

  const int TB = 256;
  auto pk = [&](const float* W, __bf16* out, int rows, int KP) {
    pack_w_kernel<<<(KP * 64 + TB - 1) / TB, TB, 0, stream>>>(W, out, rows, KP);
  };
  pk(msgW[0], msgP[0], 80, 96);
  pk(msgW[1], msgP[1], 128, 128);
  pk(msgW[2], msgP[2], 128, 128);
  for (int l = 0; l < 3; ++l) pk(aggW[l], aggP[l], 128, 128);
  pk(emlW[0], emlP[0], 144, 160);
  pk(emlW[1], emlP[1], 192, 192);
  pk(postW[0], postP[0], 64, 64);
  pk(postW[1], postP[1], 64, 64);

  // degree
  zero_kernel<<<(NN + TB - 1) / TB, TB, 0, stream>>>(deg, NN);
  deg_kernel<<<(NE + TB - 1) / TB, TB, 0, stream>>>(ei, deg, NE);

  const int EB = (NE / 16 + 7) / 8;   // edge-tile blocks (8 waves x 16 edges)
  const int NB = (NN / 16 + 7) / 8;   // node-tile blocks
  const int ZB = (NN * 64 + TB - 1) / TB;

  // ---- layer 0 ----
  zero_kernel<<<ZB, TB, 0, stream>>>(agg, NN * 64);
  msg_agg_kernel<96><<<EB, TB, 0, stream>>>(x_in, ea_in, nullptr, 16, ei, (const v16bf*)msgP[0], msgB[0], agg, NE);
  agg_update_kernel<<<NB, TB, 0, stream>>>(agg, deg, x_in, (const v16bf*)aggP[0], aggB[0], x0, NN);
  edge_mlp_kernel<160><<<EB, TB, 0, stream>>>(x0, ea_in, nullptr, 16, ei, (const v16bf*)emlP[0], emlB[0], e0b, NE);

  // ---- layer 1 ----
  zero_kernel<<<ZB, TB, 0, stream>>>(agg, NN * 64);
  msg_agg_kernel<128><<<EB, TB, 0, stream>>>(x0, nullptr, e0b, 64, ei, (const v16bf*)msgP[1], msgB[1], agg, NE);
  agg_update_kernel<<<NB, TB, 0, stream>>>(agg, deg, x0, (const v16bf*)aggP[1], aggB[1], x1, NN);
  edge_mlp_kernel<192><<<EB, TB, 0, stream>>>(x1, nullptr, e0b, 64, ei, (const v16bf*)emlP[1], emlB[1], e1b, NE);

  // ---- layer 2 (edge-MLP output is dead -> skipped) ----
  zero_kernel<<<ZB, TB, 0, stream>>>(agg, NN * 64);
  msg_agg_kernel<128><<<EB, TB, 0, stream>>>(x1, nullptr, e1b, 64, ei, (const v16bf*)msgP[2], msgB[2], agg, NE);
  agg_update_kernel<<<NB, TB, 0, stream>>>(agg, deg, x1, (const v16bf*)aggP[2], aggB[2], x0, NN);

  // ---- post MLP ----
  post_kernel<<<NB, TB, 0, stream>>>(x0, (const v16bf*)postP[0], postB[0],
                                     (const v16bf*)postP[1], postB[1], (float*)d_out, NN);
}
